// Detect_29205777613624
// MI455X (gfx1250) — compile-verified, matrix operand order checked by reading, not run on hardware
//
#include <hip/hip_runtime.h>
#include <hip/hip_bf16.h>

// ---------------------------------------------------------------------------
// Types for CDNA5 WMMA (gfx1250, wave32)
// ---------------------------------------------------------------------------
typedef __attribute__((ext_vector_type(16))) __bf16 v16bf;
typedef __attribute__((ext_vector_type(8)))  float  v8f;

union FragBF { unsigned int u[8]; v16bf v; };

__device__ __forceinline__ unsigned short f2bf(float f) {
    unsigned int x = __builtin_bit_cast(unsigned int, f);
    unsigned int r = x + 0x7FFFu + ((x >> 16) & 1u);   // round-to-nearest-even
    return (unsigned short)(r >> 16);
}
__device__ __forceinline__ float bf2f(unsigned short u) {
    unsigned int x = ((unsigned int)u) << 16;
    return __builtin_bit_cast(float, x);
}

// ---------------------------------------------------------------------------
// fp32 NCHW -> bf16 "pair-packed" layout: dword i = (b*(C/2)+ch)*S + p holds
// channels (2ch, 2ch+1) at pixel p. One b32 feeds a K-pair of a WMMA B frag.
// ---------------------------------------------------------------------------
__global__ void __launch_bounds__(256)
cvt_pack_kernel(const float* __restrict__ in, unsigned int* __restrict__ out,
                int C, int S, int B) {
    int C2h = C >> 1;
    size_t n  = (size_t)B * C2h * S;
    size_t st = (size_t)gridDim.x * blockDim.x;
    for (size_t i = (size_t)blockIdx.x * blockDim.x + threadIdx.x; i < n; i += st) {
        int p  = (int)(i % S);
        int ch = (int)((i / S) % C2h);
        int b  = (int)(i / ((size_t)C2h * S));
        const float* src = in + ((size_t)b * C + 2 * ch) * S + p;
        unsigned short lo = f2bf(src[0]);
        unsigned short hi = f2bf(src[S]);
        out[i] = (unsigned int)lo | ((unsigned int)hi << 16);
    }
}

// ---------------------------------------------------------------------------
// Pack conv weights (OIHW fp32) into WMMA A-fragment order (bf16 pairs).
// Tile order: tile = (tap*KT + kt)*MT + mt ; inside tile: dword = lane*8 + r.
// Lane L: M = mt*16 + (L&15); VGPR r: K pair k0 = kt*32 + 2(r&3)+8(L>>4)+16(r>>2)
// ---------------------------------------------------------------------------
__global__ void __launch_bounds__(256)
pack_w_kernel(const float* __restrict__ w, unsigned int* __restrict__ out,
              int Ci, int Co, int taps, int total) {
    int MT = (Co + 15) >> 4;
    int KT = Ci >> 5;
    for (int i = blockIdx.x * blockDim.x + threadIdx.x; i < total;
         i += gridDim.x * blockDim.x) {
        int tile = i >> 8, pos = i & 255;
        int lane = pos >> 3, r = pos & 7;
        int mt  = tile % MT;
        int kt  = (tile / MT) % KT;
        int tap = tile / (MT * KT);
        int half = lane >> 4;
        int m    = mt * 16 + (lane & 15);
        int k0   = kt * 32 + 2 * (r & 3) + 8 * half + 16 * (r >> 2);
        unsigned short lo = 0, hi = 0;
        if (m < Co) {
            lo = f2bf(w[((size_t)m * Ci + k0)     * taps + tap]);
            hi = f2bf(w[((size_t)m * Ci + k0 + 1) * taps + tap]);
        }
        out[i] = (unsigned int)lo | ((unsigned int)hi << 16);
    }
}

// ---------------------------------------------------------------------------
// Implicit-GEMM conv via v_wmma_f32_16x16x32_bf16.
// Block = 128 threads = 4 waves; block computes 64 co x 64 pixels.
// Per (tap, kt): the 32Kx64px B chunk is DMA'd straight into LDS with
// GLOBAL_LOAD_ASYNC_TO_LDS_B32 (ASYNCcnt, s_wait_asynccnt) -- no VGPR
// round-trip. Halo columns are zero-filled with plain ds_store (disjoint
// addresses from the async writes). Each wave loads one A fragment and
// issues 4 back-to-back WMMAs against 4 B fragments.
// epilogue 0: y=acc*scale+bias; y*sigmoid(y) -> bf16 pair-packed (hidden)
// epilogue 1: acc+bias -> fp32 into 144-ch raw tensor at channel offset
// ---------------------------------------------------------------------------
__global__ void __launch_bounds__(128)
conv_wmma_kernel(const unsigned int* __restrict__ in,     // pair-packed bf16
                 const unsigned int* __restrict__ wpack,
                 const float* __restrict__ scale,
                 const float* __restrict__ bias,
                 unsigned short* __restrict__ out_bf,     // pair-packed bf16
                 float* __restrict__ out_f32,
                 int Ci, int Co, int H, int W,
                 int taps, int MT, int co_off, int outC, int epilogue) {
    __shared__ unsigned int ldsB[1024];
    __shared__ int pixOff[64];
    __shared__ int pB[64], pP[64], pY[64], pX[64];

    const int tid  = threadIdx.x;
    const int lane = tid & 31;
    const int wave = tid >> 5;
    const int half = lane >> 4;
    const int n    = lane & 15;
    const int S    = H * W;
    const int C2h  = Ci >> 1;
    const int g0   = blockIdx.x * 64;
    const int mt   = blockIdx.y * 4 + wave;

    if (tid < 64) {                       // one-time pixel decode (divisions here only)
        int g = g0 + tid;
        int b = g / S;
        int p = g - b * S;
        int y = p / W;
        pB[tid] = b;
        pP[tid] = p;
        pY[tid] = y;
        pX[tid] = p - y * W;
    }

    v8f acc[4];
#pragma unroll
    for (int q = 0; q < 4; ++q) acc[q] = (v8f){0.f,0.f,0.f,0.f,0.f,0.f,0.f,0.f};

    const int KT = Ci >> 5;
    for (int tap = 0; tap < taps; ++tap) {
        const int dy = (taps == 9) ? (tap / 3 - 1) : 0;
        const int dx = (taps == 9) ? (tap % 3 - 1) : 0;
        if (tid < 64) {                   // per-tap shifted base offset (no div)
            int yy = pY[tid] + dy, xx = pX[tid] + dx;
            pixOff[tid] = ((unsigned)yy < (unsigned)H && (unsigned)xx < (unsigned)W)
                              ? (pB[tid] * C2h) * S + yy * W + xx
                              : -1;
        }
        __syncthreads();
        for (int kt = 0; kt < KT; ++kt) {
            // ---- async-DMA stage: 16 K-pairs x 64 px = 1024 dwords into LDS ----
            const unsigned int* chunk_base = in + (size_t)(kt * 16) * S;
#pragma unroll
            for (int it = 0; it < 8; ++it) {
                int idx = tid + it * 128;
                int kp  = idx >> 6, nn = idx & 63;
                int off = pixOff[nn];
                unsigned int lds_off = (unsigned int)(unsigned long long)&ldsB[idx];
                if (off >= 0) {
                    const unsigned int* gp = chunk_base + (size_t)off + (size_t)kp * S;
                    asm volatile("global_load_async_to_lds_b32 %0, %1, off"
                                 :: "v"(lds_off), "v"(gp)
                                 : "memory");
                } else {
                    ldsB[idx] = 0u;       // halo zero-fill (disjoint from async dst)
                }
            }
            asm volatile("s_wait_asynccnt 0x0" ::: "memory");   // my DMA done
            __syncthreads();                                    // everyone's DMA done

            if (mt < MT) {
                FragBF a;
                const unsigned int* ap =
                    wpack + (((size_t)(tap * KT + kt) * MT + mt) << 8) + lane * 8;
                __builtin_prefetch(ap + 256, 0, 1);   // global_prefetch_b8
#pragma unroll
                for (int r = 0; r < 8; ++r) a.u[r] = ap[r];
#pragma unroll
                for (int q = 0; q < 4; ++q) {
                    FragBF bm;
#pragma unroll
                    for (int r = 0; r < 8; ++r)
                        bm.u[r] = ldsB[(((r & 3) + (half << 2) + ((r >> 2) << 3)) << 6) +
                                       q * 16 + n];
                    acc[q] = __builtin_amdgcn_wmma_f32_16x16x32_bf16(
                        false, a.v, false, bm.v, (short)0, acc[q], false, false);
                }
            }
            __syncthreads();
        }
    }

    if (mt < MT) {
#pragma unroll
        for (int q = 0; q < 4; ++q) {
            int pix = q * 16 + n;
            int b = pB[pix], p = pP[pix];
#pragma unroll
            for (int r = 0; r < 8; ++r) {
                int m  = r + (half << 3);
                int co = mt * 16 + m;
                if (co < Co) {
                    float v = acc[q][r];
                    if (epilogue == 0) {
                        float y = v * scale[co] + bias[co];
                        float s = y * (1.0f / (1.0f + __expf(-y)));
                        out_bf[((size_t)(b * (Co >> 1) + (co >> 1)) * S + p) * 2 + (co & 1)] =
                            f2bf(s);
                    } else {
                        out_f32[(size_t)(b * outC + co_off + co) * S + p] = v + bias[co];
                    }
                }
            }
        }
    }
}

// ---------------------------------------------------------------------------
// Depthwise 3x3 + scale/bias + SiLU on pair-packed bf16: one thread handles a
// channel pair -> 9 dword loads, 1 dword store.
// ---------------------------------------------------------------------------
__global__ void __launch_bounds__(256)
dwconv_silu_kernel(const unsigned int* __restrict__ in,
                   const float* __restrict__ w,
                   const float* __restrict__ scale,
                   const float* __restrict__ bias,
                   unsigned int* __restrict__ out,
                   int C, int H, int W, int B) {
    int S = H * W, C2h = C >> 1;
    size_t n  = (size_t)B * C2h * S;
    size_t st = (size_t)gridDim.x * blockDim.x;
    for (size_t i = (size_t)blockIdx.x * blockDim.x + threadIdx.x; i < n; i += st) {
        int p  = (int)(i % S);
        int ch = (int)((i / S) % C2h);
        int b  = (int)(i / ((size_t)C2h * S));
        int y = p / W, x = p - (p / W) * W;
        int c0 = 2 * ch;
        const unsigned int* ip = in + (size_t)(b * C2h + ch) * S;
        float a0 = 0.f, a1 = 0.f;
#pragma unroll
        for (int ky = 0; ky < 3; ++ky)
#pragma unroll
            for (int kx = 0; kx < 3; ++kx) {
                int yy = y + ky - 1, xx = x + kx - 1;
                if ((unsigned)yy < (unsigned)H && (unsigned)xx < (unsigned)W) {
                    unsigned int v = ip[yy * W + xx];
                    a0 += bf2f((unsigned short)(v & 0xFFFFu)) * w[c0 * 9 + ky * 3 + kx];
                    a1 += bf2f((unsigned short)(v >> 16))     * w[(c0 + 1) * 9 + ky * 3 + kx];
                }
            }
        float y0 = a0 * scale[c0] + bias[c0];
        float y1 = a1 * scale[c0 + 1] + bias[c0 + 1];
        float s0 = y0 * (1.0f / (1.0f + __expf(-y0)));
        float s1 = y1 * (1.0f / (1.0f + __expf(-y1)));
        out[i] = (unsigned int)f2bf(s0) | ((unsigned int)f2bf(s1) << 16);
    }
}

// ---------------------------------------------------------------------------
// Decode: DFL softmax -> dist -> dbox*stride ; sigmoid(cls) ; row max.
// ---------------------------------------------------------------------------
__global__ void __launch_bounds__(256)
decode_kernel(const float* __restrict__ raw0, const float* __restrict__ raw1,
              const float* __restrict__ raw2,
              float* __restrict__ boxes, float* __restrict__ scores,
              float* __restrict__ rowmax, int B) {
    const int A = 8400;
    int t = blockIdx.x * blockDim.x + threadIdx.x;
    if (t >= B * A) return;
    int b = t / A, a = t % A;
    const float* raw; int S, W, p; float stride;
    if (a < 6400)      { raw = raw0; S = 6400; W = 80; p = a;        stride = 8.f;  }
    else if (a < 8000) { raw = raw1; S = 1600; W = 40; p = a - 6400; stride = 16.f; }
    else               { raw = raw2; S = 400;  W = 20; p = a - 8000; stride = 32.f; }
    const float* rb = raw + (size_t)b * 144 * S + p;

    float dist[4];
#pragma unroll
    for (int s4 = 0; s4 < 4; ++s4) {
        float v[16], m = -1e30f;
#pragma unroll
        for (int j = 0; j < 16; ++j) { v[j] = rb[(size_t)(s4 * 16 + j) * S]; m = fmaxf(m, v[j]); }
        float sum = 0.f, accd = 0.f;
#pragma unroll
        for (int j = 0; j < 16; ++j) { float e = __expf(v[j] - m); sum += e; accd += e * (float)j; }
        dist[s4] = accd / sum;
    }
    float ax = (float)(p % W) + 0.5f, ay = (float)(p / W) + 0.5f;
    float x1 = ax - dist[0], y1 = ay - dist[1];
    float x2 = ax + dist[2], y2 = ay + dist[3];
    float* bo = boxes + ((size_t)b * A + a) * 4;
    bo[0] = (x1 + x2) * 0.5f * stride;
    bo[1] = (y1 + y2) * 0.5f * stride;
    bo[2] = (x2 - x1) * stride;
    bo[3] = (y2 - y1) * stride;

    float m = -1e30f;
    float* sc = scores + ((size_t)b * A + a) * 80;
    for (int c = 0; c < 80; ++c) {
        float s = 1.0f / (1.0f + __expf(-rb[(size_t)(64 + c) * S]));
        sc[c] = s;
        m = fmaxf(m, s);
    }
    rowmax[(size_t)b * A + a] = m;
}

// ---------------------------------------------------------------------------
// Top-k stage 1: per image select top K=300 anchors by rowmax (iterative
// argmax with LDS reduction), then gather their boxes & 80-class scores.
// ---------------------------------------------------------------------------
__global__ void __launch_bounds__(256)
topk1_kernel(float* __restrict__ rowmax, const float* __restrict__ boxes,
             const float* __restrict__ scores, int A, int K,
             int* __restrict__ selIdx, float* __restrict__ selBoxes,
             float* __restrict__ selScores) {
    __shared__ float sv[256];
    __shared__ int   si[256];
    int b = blockIdx.x;
    float* rm = rowmax + (size_t)b * A;
    for (int k = 0; k < K; ++k) {
        float best = -1e30f; int bi = 0;
        for (int i = threadIdx.x; i < A; i += 256) {
            float v = rm[i];
            if (v > best) { best = v; bi = i; }
        }
        sv[threadIdx.x] = best; si[threadIdx.x] = bi;
        __syncthreads();
        for (int off = 128; off > 0; off >>= 1) {
            if (threadIdx.x < off) {
                float vo = sv[threadIdx.x + off];
                if (vo > sv[threadIdx.x] ||
                    (vo == sv[threadIdx.x] && si[threadIdx.x + off] < si[threadIdx.x])) {
                    sv[threadIdx.x] = vo; si[threadIdx.x] = si[threadIdx.x + off];
                }
            }
            __syncthreads();
        }
        if (threadIdx.x == 0) { selIdx[b * K + k] = si[0]; rm[si[0]] = -1e30f; }
        __syncthreads();
    }
    for (int t = threadIdx.x; t < K * 80; t += 256) {
        int k = t / 80, c = t % 80;
        int a = selIdx[b * K + k];
        selScores[((size_t)b * K + k) * 80 + c] = scores[((size_t)b * A + a) * 80 + c];
    }
    for (int t = threadIdx.x; t < K * 4; t += 256) {
        int k = t / 4, c = t % 4;
        int a = selIdx[b * K + k];
        selBoxes[((size_t)b * K + k) * 4 + c] = boxes[((size_t)b * A + a) * 4 + c];
    }
}

// ---------------------------------------------------------------------------
// Top-k stage 2: per image, top K of K*80 (slot,cls) scores -> output rows
// [x, y, w, h, score, cls].
// ---------------------------------------------------------------------------
__global__ void __launch_bounds__(256)
topk2_kernel(float* __restrict__ selScores, const float* __restrict__ selBoxes,
             float* __restrict__ out, int K) {
    __shared__ float sv[256];
    __shared__ int   si[256];
    int b = blockIdx.x;
    float* ss = selScores + (size_t)b * K * 80;
    const int N = K * 80;
    for (int k = 0; k < K; ++k) {
        float best = -1e30f; int bi = 0;
        for (int i = threadIdx.x; i < N; i += 256) {
            float v = ss[i];
            if (v > best) { best = v; bi = i; }
        }
        sv[threadIdx.x] = best; si[threadIdx.x] = bi;
        __syncthreads();
        for (int off = 128; off > 0; off >>= 1) {
            if (threadIdx.x < off) {
                float vo = sv[threadIdx.x + off];
                if (vo > sv[threadIdx.x] ||
                    (vo == sv[threadIdx.x] && si[threadIdx.x + off] < si[threadIdx.x])) {
                    sv[threadIdx.x] = vo; si[threadIdx.x] = si[threadIdx.x + off];
                }
            }
            __syncthreads();
        }
        if (threadIdx.x == 0) {
            int idx = si[0]; float val = sv[0];
            int slot = idx / 80, cls = idx % 80;
            float* o = out + ((size_t)b * K + k) * 6;
            const float* bx = selBoxes + ((size_t)b * K + slot) * 4;
            o[0] = bx[0]; o[1] = bx[1]; o[2] = bx[2]; o[3] = bx[3];
            o[4] = val;   o[5] = (float)cls;
            ss[idx] = -1e30f;
        }
        __syncthreads();
    }
}

// ---------------------------------------------------------------------------
// Host orchestration
// ---------------------------------------------------------------------------
static inline size_t align256(size_t x) { return (x + 255) & ~(size_t)255; }

extern "C" void kernel_launch(void* const* d_in, const int* in_sizes, int n_in,
                              void* d_out, int out_size, void* d_ws, size_t ws_size,
                              hipStream_t stream) {
    (void)in_sizes; (void)n_in; (void)out_size; (void)ws_size;

    const int B = 16, K = 300, A = 8400, NO = 144, C3 = 256, C2 = 64, NC = 80;
    const int CHs[3] = {256, 512, 1024};
    const int HHs[3] = {80, 40, 20};
    const int Ss[3]  = {6400, 1600, 400};

    const float* x[3] = {(const float*)d_in[0], (const float*)d_in[1], (const float*)d_in[2]};
    auto P = [&](int lvl, int j) { return (const float*)d_in[3 + lvl * 22 + j]; };

    char* ws = (char*)d_ws;
    size_t off = 0;
    auto alloc = [&](size_t bytes) { void* p = ws + off; off = align256(off + bytes); return p; };

    float* raw[3]; unsigned int* xb[3];
    for (int l = 0; l < 3; ++l) {
        raw[l] = (float*)alloc((size_t)B * NO * Ss[l] * 4);
        xb[l]  = (unsigned int*)alloc((size_t)B * CHs[l] * Ss[l] * 2);
    }
    unsigned int* t0   = (unsigned int*)alloc((size_t)B * C2 * 6400 * 2);
    unsigned int* t1   = (unsigned int*)alloc((size_t)B * C2 * 6400 * 2);
    unsigned int* big0 = (unsigned int*)alloc((size_t)B * 256 * 6400 * 2);
    unsigned int* big1 = (unsigned int*)alloc((size_t)B * 256 * 6400 * 2);

    auto packed_dwords = [](int taps, int Ci, int Co) {
        return (size_t)taps * (size_t)(Ci / 32) * (size_t)((Co + 15) / 16) * 256;
    };
    unsigned int* wp[3][6];
    for (int l = 0; l < 3; ++l) {
        int cin = CHs[l];
        wp[l][0] = (unsigned int*)alloc(packed_dwords(9, cin, C2) * 4);
        wp[l][1] = (unsigned int*)alloc(packed_dwords(9, C2, C2) * 4);
        wp[l][2] = (unsigned int*)alloc(packed_dwords(1, C2, C2) * 4);
        wp[l][3] = (unsigned int*)alloc(packed_dwords(1, cin, C3) * 4);
        wp[l][4] = (unsigned int*)alloc(packed_dwords(1, C3, C3) * 4);
        wp[l][5] = (unsigned int*)alloc(packed_dwords(1, C3, NC) * 4);
    }
    float* boxes     = (float*)alloc((size_t)B * A * 4 * 4);
    float* scores    = (float*)alloc((size_t)B * A * 80 * 4);
    float* rowmax    = (float*)alloc((size_t)B * A * 4);
    int*   selIdx    = (int*)alloc((size_t)B * K * 4);
    float* selBoxes  = (float*)alloc((size_t)B * K * 4 * 4);
    float* selScores = (float*)alloc((size_t)B * K * 80 * 4);

    auto launch_cvt = [&](const float* src, unsigned int* dst, int C, int S) {
        size_t n = (size_t)B * (C >> 1) * S;
        int blocks = (int)((n + 255) / 256); if (blocks > 8192) blocks = 8192;
        cvt_pack_kernel<<<blocks, 256, 0, stream>>>(src, dst, C, S, B);
    };
    auto launch_pack = [&](const float* w, unsigned int* outp, int taps, int Ci, int Co) {
        int total = (int)packed_dwords(taps, Ci, Co);
        int blocks = (total + 255) / 256; if (blocks > 2048) blocks = 2048;
        pack_w_kernel<<<blocks, 256, 0, stream>>>(w, outp, Ci, Co, taps, total);
    };
    auto launch_conv = [&](const unsigned int* in, const unsigned int* wpk,
                           const float* scale, const float* bias,
                           unsigned int* outb, float* outf,
                           int Ci, int Co, int H, int W, int taps,
                           int co_off, int outC, int epi) {
        int S = H * W;
        int PT = (B * S) / 64;            // B*S is a multiple of 64 at all levels
        int MT = (Co + 15) / 16;
        dim3 grid(PT, (MT + 3) / 4);
        conv_wmma_kernel<<<grid, 128, 0, stream>>>(in, wpk, scale, bias,
                                                   (unsigned short*)outb, outf,
                                                   Ci, Co, H, W, taps, MT, co_off, outC, epi);
    };
    auto launch_dw = [&](const unsigned int* in, const float* w, const float* scale,
                         const float* bias, unsigned int* outp, int C, int H, int W) {
        size_t n = (size_t)B * (C >> 1) * H * W;
        int blocks = (int)((n + 255) / 256); if (blocks > 8192) blocks = 8192;
        dwconv_silu_kernel<<<blocks, 256, 0, stream>>>(in, w, scale, bias, outp, C, H, W, B);
    };

    // ---- convert inputs to pair-packed bf16 (fp32 read exactly once) ----
    for (int l = 0; l < 3; ++l)
        launch_cvt(x[l], xb[l], CHs[l], Ss[l]);

    // ---- pack all weights into WMMA A-fragment order ----
    for (int l = 0; l < 3; ++l) {
        int cin = CHs[l];
        launch_pack(P(l, 0),  wp[l][0], 9, cin, C2);
        launch_pack(P(l, 3),  wp[l][1], 9, C2, C2);
        launch_pack(P(l, 6),  wp[l][2], 1, C2, C2);
        launch_pack(P(l, 11), wp[l][3], 1, cin, C3);
        launch_pack(P(l, 17), wp[l][4], 1, C3, C3);
        launch_pack(P(l, 20), wp[l][5], 1, C3, NC);
    }

    // ---- per-level head ----
    for (int l = 0; l < 3; ++l) {
        int cin = CHs[l], H = HHs[l], W = HHs[l];
        // cv2 branch: 3x3 -> 3x3 -> 1x1 (+bb) into raw[:, 0:64]
        launch_conv(xb[l], wp[l][0], P(l, 1), P(l, 2), t0, nullptr, cin, C2, H, W, 9, 0, 0, 0);
        launch_conv(t0,    wp[l][1], P(l, 4), P(l, 5), t1, nullptr, C2,  C2, H, W, 9, 0, 0, 0);
        launch_conv(t1,    wp[l][2], nullptr, P(l, 7), nullptr, raw[l], C2, C2, H, W, 1, 0, NO, 1);
        // cv3 branch: dw -> pw -> dw -> pw -> 1x1 (+bc) into raw[:, 64:144]
        launch_dw(xb[l], P(l, 8),  P(l, 9),  P(l, 10), big0, cin, H, W);
        launch_conv(big0, wp[l][3], P(l, 12), P(l, 13), big1, nullptr, cin, C3, H, W, 1, 0, 0, 0);
        launch_dw(big1, P(l, 14), P(l, 15), P(l, 16), big0, C3, H, W);
        launch_conv(big0, wp[l][4], P(l, 18), P(l, 19), big1, nullptr, C3, C3, H, W, 1, 0, 0, 0);
        launch_conv(big1, wp[l][5], nullptr, P(l, 21), nullptr, raw[l], C3, NC, H, W, 1, 64, NO, 1);
    }

    // ---- decode + two-stage top-k ----
    {
        int total = B * A;
        decode_kernel<<<(total + 255) / 256, 256, 0, stream>>>(raw[0], raw[1], raw[2],
                                                               boxes, scores, rowmax, B);
        topk1_kernel<<<B, 256, 0, stream>>>(rowmax, boxes, scores, A, K,
                                            selIdx, selBoxes, selScores);
        topk2_kernel<<<B, 256, 0, stream>>>(selScores, selBoxes, (float*)d_out, K);
    }
}
